// GraphSAGE_56075093017243
// MI455X (gfx1250) — matrix operand order repaired and checked by reading, not compiled, not run on hardware
//
#include <hip/hip_runtime.h>
#include <hip/hip_bf16.h>

// ---------------------------------------------------------------------------
// GraphSAGE on MI455X (gfx1250, wave32).
//  - Edge aggregation: native f32 global atomics (all tensors are L2-resident
//    in the 192MB L2, so random scatter/gather traffic stays on-chip).
//  - Linear layers: fused [x | neigh_mean] @ [Ws | Wn].T as ONE f16 WMMA GEMM
//    per layer (v_wmma_f32_16x16x32_f16, f32 accumulate), LDS-tiled.
//  - Weight tile staged into LDS by the Tensor Data Mover (tensor_load_to_lds)
//    with hardware row padding; falls back to a manual copy if the builtin is
//    unavailable.
// ---------------------------------------------------------------------------

#ifndef __has_builtin
#define __has_builtin(x) 0
#endif

#if defined(__AMDGCN__) && __has_builtin(__builtin_amdgcn_tensor_load_to_lds) && \
    __has_builtin(__builtin_amdgcn_s_wait_tensorcnt)
#define USE_TDM 1
#endif

typedef __attribute__((ext_vector_type(16))) _Float16 v16h;
typedef __attribute__((ext_vector_type(8)))  _Float16 v8h;
typedef __attribute__((ext_vector_type(8)))  float    v8f;

#if defined(USE_TDM)
typedef unsigned int tdm_u32x4 __attribute__((ext_vector_type(4)));
typedef int          tdm_i32x8 __attribute__((ext_vector_type(8)));
typedef int          tdm_i32x4 __attribute__((ext_vector_type(4)));
#endif

// ---------------------------- utility kernels ------------------------------

__global__ void fzero_kernel(float* __restrict__ p, long long n) {
    long long i = blockIdx.x * (long long)blockDim.x + threadIdx.x;
    long long stride = (long long)gridDim.x * blockDim.x;
    for (; i < n; i += stride) p[i] = 0.0f;
}

__global__ void degree_kernel(const int* __restrict__ dst,
                              float* __restrict__ deg, int e) {
    int i = blockIdx.x * blockDim.x + threadIdx.x;
    if (i < e) unsafeAtomicAdd(&deg[dst[i]], 1.0f);
}

// Gather x[src], scatter-add into S[dst]; one thread per (edge, 4-float chunk).
template <int D>
__global__ void scatter_kernel(const float* __restrict__ X,
                               const int* __restrict__ src,
                               const int* __restrict__ dst,
                               float* __restrict__ S, int e) {
    constexpr int CH = D / 4;
    long long idx = blockIdx.x * (long long)blockDim.x + threadIdx.x;
    long long total = (long long)e * CH;
    if (idx >= total) return;
    int eid = (int)(idx / CH);
    int c   = (int)(idx % CH) * 4;
    int s = src[eid], d = dst[eid];
    const float4 v = *(const float4*)&X[(long long)s * D + c];
    float* p = &S[(long long)d * D + c];
    unsafeAtomicAdd(p + 0, v.x);
    unsafeAtomicAdd(p + 1, v.y);
    unsafeAtomicAdd(p + 2, v.z);
    unsafeAtomicAdd(p + 3, v.w);
}

// Pack [Ws | Wn] (each [dout][DIN] f32) into f16 [dout][2*DIN] row-major.
template <int DIN>
__global__ void pack_weights_kernel(const float* __restrict__ Ws,
                                    const float* __restrict__ Wn,
                                    _Float16* __restrict__ Wc, int dout) {
    int i = blockIdx.x * blockDim.x + threadIdx.x;
    int total = dout * 2 * DIN;
    if (i >= total) return;
    int o = i / (2 * DIN), k = i % (2 * DIN);
    float v = (k < DIN) ? Ws[o * DIN + k] : Wn[o * DIN + (k - DIN)];
    Wc[i] = (_Float16)v;
}

__global__ void convert_f16_kernel(const float* __restrict__ W,
                                   _Float16* __restrict__ Wh, int total) {
    int i = blockIdx.x * blockDim.x + threadIdx.x;
    if (i < total) Wh[i] = (_Float16)W[i];
}

__global__ void bias_sum_kernel(const float* __restrict__ a,
                                const float* __restrict__ b,
                                float* __restrict__ c, int n) {
    int i = blockIdx.x * blockDim.x + threadIdx.x;
    if (i < n) c[i] = a[i] + b[i];
}

// ------------------------- fused WMMA SAGE layer ---------------------------
// OUT[n, DOUT] = act( [X | NS/deg] (n x K) @ W.T + bias )
// Block: 256 threads = 8 waves; each wave owns a 16-row tile -> 128 rows/block.
// A and W live in LDS as f16 with +8-half row padding (bank-conflict free).
template <int K, int DOUT, bool RELU, bool HASNEIGH>
__global__ void sage_gemm_kernel(const float* __restrict__ X,
                                 const float* __restrict__ NS,
                                 const float* __restrict__ deg,
                                 const _Float16* __restrict__ W,
                                 const float* __restrict__ bias,
                                 float* __restrict__ OUT, int n) {
    constexpr int ROWS = 128;
    constexpr int DIN  = HASNEIGH ? (K / 2) : K;
    constexpr int LDA  = K + 8;           // padded LDS row stride (halves)
    constexpr int CT   = DOUT / 16;       // output column tiles per wave

    extern __shared__ _Float16 smem[];
    _Float16* sA = smem;                  // [ROWS][LDA]
    _Float16* sW = smem + ROWS * LDA;     // [DOUT][LDA]

    const int tid  = threadIdx.x;
    const int row0 = blockIdx.x * ROWS;

    // ---- stage weights [DOUT][K] f16 -> LDS [DOUT][LDA] ----
#if defined(USE_TDM)
    // Tensor Data Mover: 2D f16 tensor, hardware inserts the 8-half (4-DWORD)
    // row pad after every K/2 DWORDs. Issued by wave 0 only; TENSORcnt waited
    // before the workgroup barrier.
    if (tid < 32) {
        const unsigned lds_off = (unsigned)(unsigned long long)(uintptr_t)sW;
        const unsigned long long ga = (unsigned long long)(uintptr_t)W;
        constexpr unsigned PAD_INTERVAL = (K / 2 == 64) ? 5u : 6u; // 64/128 DW
        constexpr unsigned PAD_AMOUNT   = 3u;                      // 4 DWORDs

        tdm_u32x4 g0;
        g0[0] = 1u;                                   // count=1, user descriptor
        g0[1] = lds_off;                              // LDS byte address
        g0[2] = (unsigned)(ga & 0xFFFFFFFFu);         // global_addr[31:0]
        g0[3] = (unsigned)((ga >> 32) & 0x1FFFFFFu)   // global_addr[56:32]
                | (2u << 30);                         // type = 2 ("image")

        tdm_i32x8 g1;
        g1[0] = (int)((1u << 16)                      // data_size = 2 bytes
                    | (1u << 20)                      // pad_enable
                    | (PAD_INTERVAL << 22)
                    | (PAD_AMOUNT << 25));
        g1[1] = (int)(((unsigned)K & 0xFFFFu) << 16); // tensor_dim0[15:0]
        g1[2] = (int)((((unsigned)K >> 16) & 0xFFFFu)
                    | (((unsigned)DOUT & 0xFFFFu) << 16)); // tensor_dim1[15:0]
        g1[3] = (int)((((unsigned)DOUT >> 16) & 0xFFFFu)
                    | (((unsigned)K & 0xFFFFu) << 16));    // tile_dim0
        g1[4] = (int)((unsigned)DOUT & 0xFFFFu);      // tile_dim1 (tile_dim2=0)
        g1[5] = (int)K;                               // tensor_dim0_stride[31:0]
        g1[6] = 0;                                    // stride hi / dim1_stride
        g1[7] = 0;

        tdm_i32x4 g2 = {0, 0, 0, 0};
        tdm_i32x4 g3 = {0, 0, 0, 0};
#if __has_include(<hip/amd_detail/amd_gfx1250_TDM.h>)
        tdm_i32x8 g4 = {0, 0, 0, 0, 0, 0, 0, 0};
        __builtin_amdgcn_tensor_load_to_lds(g0, g1, g2, g3, g4, 0);
#else
        __builtin_amdgcn_tensor_load_to_lds(g0, g1, g2, g3, 0);
#endif
        __builtin_amdgcn_s_wait_tensorcnt(0);
    }
#else
    for (int i = tid; i < DOUT * (K / 2); i += blockDim.x) {
        int o = i / (K / 2), kp = i - o * (K / 2);
        ((unsigned int*)&sW[o * LDA])[kp] = ((const unsigned int*)W)[i];
    }
#endif

    // ---- stage A tile: f32 -> f16, fold in neighbor-mean division ----
    for (int i = tid; i < ROWS * K; i += blockDim.x) {
        int r = i / K, k = i - r * K;
        int node = row0 + r;
        float v = 0.0f;
        if (node < n) {
            if constexpr (HASNEIGH) {
                if (k < DIN) {
                    v = X[(long long)node * DIN + k];
                } else {
                    float d = deg[node];
                    v = NS[(long long)node * DIN + (k - DIN)] / fmaxf(d, 1.0f);
                }
            } else {
                v = X[(long long)node * K + k];
            }
        }
        sA[r * LDA + k] = (_Float16)v;
    }
    __syncthreads();

    const int wave = tid >> 5;            // 0..7
    const int lane = tid & 31;
    const int m    = lane & 15;           // row-in-tile (A) / column (B,C)
    const int hi   = lane >> 4;           // lane half selects K sub-range
    const int rowA = wave * 16 + m;

    v8f acc[CT] = {};

    union Frag { v16h v; v8h h[2]; };

    for (int k0 = 0; k0 < K; k0 += 32) {
        // A fragment (16x32 f16): lane half hi selects
        // K = k0 + hi*8 + {0..7} and k0 + hi*8 + 16 + {0..7}.
        Frag a;
        const int ka = k0 + hi * 8;
        a.h[0] = *(const v8h*)&sA[rowA * LDA + ka];
        a.h[1] = *(const v8h*)&sA[rowA * LDA + ka + 16];

        // Batch all CT B-fragment loads so their DS traffic overlaps and the
        // WMMA chain needs only a single dscnt wait per k-step.
        Frag b[CT];
        const int kb = k0 + hi * 8;
#pragma unroll
        for (int t = 0; t < CT; ++t) {
            const int ncol = t * 16 + m;  // B column per lane
            b[t].h[0] = *(const v8h*)&sW[ncol * LDA + kb];
            b[t].h[1] = *(const v8h*)&sW[ncol * LDA + kb + 16];
        }
#pragma unroll
        for (int t = 0; t < CT; ++t) {
            acc[t] = __builtin_amdgcn_wmma_f32_16x16x32_f16(
                false, a.v, false, b[t].v, (short)0, acc[t], false, false);
        }
    }

    // ---- epilogue: bias (+ ReLU), write f32 output ----
    const int mbase = hi * 8;             // C layout: lanes 16-31 hold M=8..15
#pragma unroll
    for (int t = 0; t < CT; ++t) {
        const int ncol = t * 16 + m;
        const float bb = bias[ncol];
#pragma unroll
        for (int r = 0; r < 8; ++r) {
            int node = row0 + wave * 16 + mbase + r;
            if (node < n) {
                float v = acc[t][r] + bb;
                if constexpr (RELU) v = fmaxf(v, 0.0f);
                OUT[(long long)node * DOUT + ncol] = v;
            }
        }
    }
}

// ------------------------------- launcher ----------------------------------

extern "C" void kernel_launch(void* const* d_in, const int* in_sizes, int n_in,
                              void* d_out, int out_size, void* d_ws, size_t ws_size,
                              hipStream_t stream) {
    constexpr int IN_DIM = 64, HIDDEN = 128, OUT_DIM = 64;

    const float* x   = (const float*)d_in[0];
    const int*   ei  = (const int*)d_in[1];
    const float* Ws1 = (const float*)d_in[2];
    const float* bs1 = (const float*)d_in[3];
    const float* Wn1 = (const float*)d_in[4];
    const float* bn1 = (const float*)d_in[5];
    const float* Ws2 = (const float*)d_in[6];
    const float* bs2 = (const float*)d_in[7];
    const float* Wn2 = (const float*)d_in[8];
    const float* bn2 = (const float*)d_in[9];
    const float* Wo  = (const float*)d_in[10];
    const float* bo  = (const float*)d_in[11];

    const int N = in_sizes[0] / IN_DIM;
    const int E = in_sizes[1] / 2;
    const int* src = ei;
    const int* dst = ei + E;

    // ---- workspace carve-out (256B aligned) ----
    char*  ws  = (char*)d_ws;
    size_t off = 0;
    auto alloc = [&](size_t bytes) -> void* {
        void* p = ws + off;
        off = (off + bytes + 255) & ~(size_t)255;
        return p;
    };
    float*     deg  = (float*)alloc((size_t)N * sizeof(float));
    float*     sum1 = (float*)alloc((size_t)N * IN_DIM * sizeof(float));
    float*     sum2 = (float*)alloc((size_t)N * HIDDEN * sizeof(float));
    float*     h1   = (float*)alloc((size_t)N * HIDDEN * sizeof(float));
    float*     h2   = (float*)alloc((size_t)N * HIDDEN * sizeof(float));
    _Float16*  Wc1  = (_Float16*)alloc((size_t)HIDDEN * 2 * IN_DIM * sizeof(_Float16));
    _Float16*  Wc2  = (_Float16*)alloc((size_t)HIDDEN * 2 * HIDDEN * sizeof(_Float16));
    _Float16*  Wo16 = (_Float16*)alloc((size_t)OUT_DIM * HIDDEN * sizeof(_Float16));
    float*     b1   = (float*)alloc(HIDDEN * sizeof(float));
    float*     b2   = (float*)alloc(HIDDEN * sizeof(float));

    const int TB = 256;
    auto cdiv = [](long long a, long long b) { return (int)((a + b - 1) / b); };

    // 1) zero accumulation scratch
    fzero_kernel<<<4096, TB, 0, stream>>>(deg,  (long long)N);
    fzero_kernel<<<4096, TB, 0, stream>>>(sum1, (long long)N * IN_DIM);
    fzero_kernel<<<4096, TB, 0, stream>>>(sum2, (long long)N * HIDDEN);

    // 2) degrees + weight/bias prep
    degree_kernel<<<cdiv(E, TB), TB, 0, stream>>>(dst, deg, E);
    pack_weights_kernel<IN_DIM><<<cdiv(HIDDEN * 2 * IN_DIM, TB), TB, 0, stream>>>(Ws1, Wn1, Wc1, HIDDEN);
    pack_weights_kernel<HIDDEN><<<cdiv(HIDDEN * 2 * HIDDEN, TB), TB, 0, stream>>>(Ws2, Wn2, Wc2, HIDDEN);
    convert_f16_kernel<<<cdiv(OUT_DIM * HIDDEN, TB), TB, 0, stream>>>(Wo, Wo16, OUT_DIM * HIDDEN);
    bias_sum_kernel<<<1, HIDDEN, 0, stream>>>(bs1, bn1, b1, HIDDEN);
    bias_sum_kernel<<<1, HIDDEN, 0, stream>>>(bs2, bn2, b2, HIDDEN);

    const int gemm_blocks = cdiv(N, 128);

    // 3) layer 1: scatter + fused WMMA GEMM (K = 2*64 = 128)
    scatter_kernel<IN_DIM><<<cdiv((long long)E * (IN_DIM / 4), TB), TB, 0, stream>>>(x, src, dst, sum1, E);
    {
        constexpr int K = 2 * IN_DIM;
        size_t shmem = (size_t)(128 * (K + 8) + HIDDEN * (K + 8)) * sizeof(_Float16);
        sage_gemm_kernel<K, HIDDEN, true, true><<<gemm_blocks, TB, shmem, stream>>>(
            x, sum1, deg, Wc1, b1, h1, N);
    }

    // 4) layer 2: scatter + fused WMMA GEMM (K = 2*128 = 256)
    scatter_kernel<HIDDEN><<<cdiv((long long)E * (HIDDEN / 4), TB), TB, 0, stream>>>(h1, src, dst, sum2, E);
    {
        constexpr int K = 2 * HIDDEN;
        size_t shmem = (size_t)(128 * (K + 8) + HIDDEN * (K + 8)) * sizeof(_Float16);
        sage_gemm_kernel<K, HIDDEN, true, true><<<gemm_blocks, TB, shmem, stream>>>(
            h1, sum2, deg, Wc2, b2, h2, N);
    }

    // 5) output projection: WMMA GEMM (K = 128, DOUT = 64, no ReLU)
    {
        constexpr int K = HIDDEN;
        size_t shmem = (size_t)(128 * (K + 8) + OUT_DIM * (K + 8)) * sizeof(_Float16);
        sage_gemm_kernel<K, OUT_DIM, false, false><<<gemm_blocks, TB, shmem, stream>>>(
            h2, nullptr, nullptr, Wo16, bo, (float*)d_out, N);
    }
}
